// MiniCDDDInference_73821897884126
// MI455X (gfx1250) — compile-verified, matrix operand order checked by reading, not run on hardware
//
#include <hip/hip_runtime.h>
#include <hip/hip_bf16.h>

typedef __attribute__((ext_vector_type(16))) __bf16 v16bf;
typedef __attribute__((ext_vector_type(8)))  float  v8f;
typedef unsigned short u16;

static constexpr int Bsz = 256, Tmax = 128, EMBD = 32, LAT = 512;
static constexpr int S0 = 512, S1 = 1024, S2 = 2048, SSUM = 3584;

__device__ __forceinline__ u16 f2bf(float x) {
  unsigned u = __float_as_uint(x);
  u += 0x7FFFu + ((u >> 16) & 1u);          // round-to-nearest-even
  return (u16)(u >> 16);
}

__device__ __forceinline__ float fast_sigmoid(float x) {
  return __builtin_amdgcn_rcpf(1.0f + __expf(-x));       // v_rcp_f32 + v_exp_f32
}
__device__ __forceinline__ float fast_tanh(float x) {
  // tanh(x) = 2*sigmoid(2x) - 1
  return fmaf(2.0f, __builtin_amdgcn_rcpf(1.0f + __expf(-2.0f * x)), -1.0f);
}

union FragU { v16bf v; float4 f4[2]; };

// Load one 16x32 bf16 WMMA operand fragment at base + element offset.
// Indexing directly off the kernarg base keeps loads on the GLOBAL path
// (addrspace inference), avoiding flat_load's dual LOADcnt/DScnt cost.
// 16-bit operand layout: lane<16 holds K {0..7,16..23}; lane>=16 holds
// {8..15,24..31} -> two b128 loads at +0 and +32 bytes.
__device__ __forceinline__ v16bf frag_at(const u16* __restrict__ base, long off) {
  const float4* p = reinterpret_cast<const float4*>(base + off);
  FragU u;
  u.f4[0] = p[0];
  u.f4[1] = p[2];
  return u.v;
}

struct GemmArgs {
  const u16* Ax; int ldax; int K1;     // x segment of concat input
  const u16* Ah; int ldah; int K2;     // h (or r*h) segment
  const u16* BT;                        // weights bf16, [N, K1+K2] row-major
  const float* bias;
  int outs;                             // hidden size of this layer
  int ldh;                              // stride of f32 state buffer (SSUM)
  const float* HfR;                     // gates: masked h (read)
  float* HfW;                           // cand: h (read+write, in place)
  float* Z;                             // z gate scratch [B, outs]
  u16* RHb;                             // r*h bf16 scratch [B, outs]
  u16* Sb;                              // masked bf16 state (this layer's cols), ld SSUM
  u16* Nb;                              // unmasked n bf16 -> next layer input (may be null)
  const int* lens; int t;
  float* Out;                           // final output [B, LAT]
};

// MODE: 0 = gates (sigmoid, emit r*h and z), 1 = candidate (tanh + GRU blend + mask),
//       2 = final projection (tanh -> Out)
template <int MODE>
__global__ __launch_bounds__(256) void gru_gemm(GemmArgs ga) {
  const int lane = threadIdx.x & 31;
  const int wave = threadIdx.x >> 5;
  const int wm = wave & 3;              // 4 waves along M (256 rows)
  const int wn = wave >> 2;             // 2 waves along N
  const int m0 = wm * 64;
  const int n0 = blockIdx.x * 128 + wn * 64;

  v8f acc[4][4] = {};
  const int Ktot = ga.K1 + ga.K2;
  const int la = lane & 15;             // row-within-tile (A: M, B: N)
  const int lk = (lane >> 4) << 3;      // K sub-offset per half-wave

  // Integer element-offset induction variables (+32 elements / 64 B per step).
  long ob[4];
#pragma unroll
  for (int nt = 0; nt < 4; ++nt)
    ob[nt] = (long)(n0 + nt * 16 + la) * Ktot + lk;
  long oa[4];
#pragma unroll
  for (int mt = 0; mt < 4; ++mt)
    oa[mt] = (long)(m0 + mt * 16 + la) * ga.ldax + lk;

  auto kstep = [&](const u16* __restrict__ abase) {
    v16bf af[4], bv[4];
#pragma unroll
    for (int mt = 0; mt < 4; ++mt) { af[mt] = frag_at(abase, oa[mt]); oa[mt] += 32; }
#pragma unroll
    for (int nt = 0; nt < 4; ++nt) { bv[nt] = frag_at(ga.BT, ob[nt]); ob[nt] += 32; }
#pragma unroll
    for (int mt = 0; mt < 4; ++mt)
#pragma unroll
      for (int nt = 0; nt < 4; ++nt)
        acc[mt][nt] = __builtin_amdgcn_wmma_f32_16x16x32_bf16(
            false, af[mt], false, bv[nt], (short)0, acc[mt][nt], false, false);
  };

  for (int s = ga.K1 >> 5; s > 0; --s) kstep(ga.Ax);
#pragma unroll
  for (int mt = 0; mt < 4; ++mt)
    oa[mt] = (long)(m0 + mt * 16 + la) * ga.ldah + lk;
  for (int s = ga.K2 >> 5; s > 0; --s) kstep(ga.Ah);

  // C layout: element i of v8f = C[m0 + mt*16 + 8*(lane>=16) + i][n0 + nt*16 + lane%16]
  const int colL = lane & 15;
  const int rowH = (lane >> 4) << 3;
#pragma unroll
  for (int mt = 0; mt < 4; ++mt) {
#pragma unroll
    for (int nt = 0; nt < 4; ++nt) {
      const int col = n0 + nt * 16 + colL;
      const float bc = ga.bias[col];
#pragma unroll
      for (int i = 0; i < 8; ++i) {
        const int row = m0 + mt * 16 + rowH + i;
        const float v = acc[mt][nt][i] + bc;
        if (MODE == 0) {
          const float g = fast_sigmoid(v);
          if (col < ga.outs) {            // r gate -> r*h (bf16, next GEMM's A)
            const float rh = g * ga.HfR[(size_t)row * ga.ldh + col];
            ga.RHb[(size_t)row * ga.outs + col] = f2bf(rh);
          } else {                        // z gate
            ga.Z[(size_t)row * ga.outs + (col - ga.outs)] = g;
          }
        } else if (MODE == 1) {
          const float c = fast_tanh(v);
          const float z = ga.Z[(size_t)row * ga.outs + col];
          const float hold = ga.HfW[(size_t)row * ga.ldh + col];
          const float hn = z * hold + (1.0f - z) * c;      // unmasked new state
          const bool msk = ga.t < ga.lens[row];
          const float hm = msk ? hn : hold;                 // masked state
          ga.HfW[(size_t)row * ga.ldh + col] = hm;
          ga.Sb[(size_t)row * SSUM + col] = f2bf(hm);
          if (ga.Nb) ga.Nb[(size_t)row * ga.outs + col] = f2bf(hn);
        } else {
          ga.Out[(size_t)row * LAT + col] = fast_tanh(v);
        }
      }
    }
  }
}

// fp32 [K,N] -> bf16 transposed [N,K]
__global__ void cvtT(const float* __restrict__ in, u16* __restrict__ out,
                     int K, int N) {
  long i = (long)blockIdx.x * 256 + threadIdx.x;
  if (i >= (long)K * N) return;
  int k = (int)(i / N), n = (int)(i % N);
  out[(size_t)n * K + k] = f2bf(in[i]);
}

// fp32 -> bf16, same layout (W is already [LAT, SSUM] = [N, K] for out = state @ W^T)
__global__ void cvtN(const float* __restrict__ in, u16* __restrict__ out, long n) {
  long i = (long)blockIdx.x * 256 + threadIdx.x;
  if (i < n) out[i] = f2bf(in[i]);
}

// embedding lookup -> bf16 [B, T*EMB]
__global__ void embedK(const int* __restrict__ seqs, const float* __restrict__ emb,
                       u16* __restrict__ xe, long n) {
  long i = (long)blockIdx.x * 256 + threadIdx.x;
  if (i >= n) return;
  int b = (int)(i / (Tmax * EMBD));
  int r = (int)(i % (Tmax * EMBD));
  int t = r / EMBD, e = r % EMBD;
  xe[i] = f2bf(emb[seqs[b * Tmax + t] * EMBD + e]);
}

extern "C" void kernel_launch(void* const* d_in, const int* in_sizes, int n_in,
                              void* d_out, int out_size, void* d_ws, size_t ws_size,
                              hipStream_t stream) {
  (void)in_sizes; (void)n_in; (void)out_size; (void)ws_size;
  const int*   seqs = (const int*)d_in[0];
  const int*   lens = (const int*)d_in[1];
  const float* emb  = (const float*)d_in[2];
  const float* Kg[3] = {(const float*)d_in[3], (const float*)d_in[7],  (const float*)d_in[11]};
  const float* bg[3] = {(const float*)d_in[4], (const float*)d_in[8],  (const float*)d_in[12]};
  const float* Kc[3] = {(const float*)d_in[5], (const float*)d_in[9],  (const float*)d_in[13]};
  const float* bc[3] = {(const float*)d_in[6], (const float*)d_in[10], (const float*)d_in[14]};
  const float* W  = (const float*)d_in[15];
  const float* bb = (const float*)d_in[16];
  float* Out = (float*)d_out;

  const int ins[3]  = {EMBD, S0, S1};
  const int outs[3] = {S0, S1, S2};
  const int soff[3] = {0, S0, S0 + S1};

  char* ws = (char*)d_ws;
  size_t off = 0;
  auto alloc = [&](size_t bytes) -> void* {
    void* p = ws + off;
    off = (off + bytes + 255) & ~(size_t)255;
    return p;
  };
  u16* KgT[3]; u16* KcT[3];
  for (int l = 0; l < 3; ++l) {
    KgT[l] = (u16*)alloc((size_t)(ins[l] + outs[l]) * 2 * outs[l] * 2);
    KcT[l] = (u16*)alloc((size_t)(ins[l] + outs[l]) * outs[l] * 2);
  }
  u16*   WT  = (u16*)alloc((size_t)LAT * SSUM * 2);
  u16*   Xe  = (u16*)alloc((size_t)Bsz * Tmax * EMBD * 2);
  float* Hf  = (float*)alloc((size_t)Bsz * SSUM * 4);
  u16*   Sb  = (u16*)alloc((size_t)Bsz * SSUM * 2);
  u16*   Nb0 = (u16*)alloc((size_t)Bsz * S0 * 2);
  u16*   Nb1 = (u16*)alloc((size_t)Bsz * S1 * 2);
  u16*   RHb = (u16*)alloc((size_t)Bsz * S2 * 2);
  float* Zb  = (float*)alloc((size_t)Bsz * S2 * 4);

  // One-time (per call) weight conversion: bf16, transposed to [N,K].
  for (int l = 0; l < 3; ++l) {
    const int K = ins[l] + outs[l];
    long ng = (long)K * 2 * outs[l];
    cvtT<<<(ng + 255) / 256, 256, 0, stream>>>(Kg[l], KgT[l], K, 2 * outs[l]);
    long nc = (long)K * outs[l];
    cvtT<<<(nc + 255) / 256, 256, 0, stream>>>(Kc[l], KcT[l], K, outs[l]);
  }
  { long n = (long)LAT * SSUM; cvtN<<<(n + 255) / 256, 256, 0, stream>>>(W, WT, n); }
  { long n = (long)Bsz * Tmax * EMBD; embedK<<<(n + 255) / 256, 256, 0, stream>>>(seqs, emb, Xe, n); }
  hipMemsetAsync(Hf, 0, (size_t)Bsz * SSUM * 4, stream);
  hipMemsetAsync(Sb, 0, (size_t)Bsz * SSUM * 2, stream);

  for (int t = 0; t < Tmax; ++t) {
    for (int l = 0; l < 3; ++l) {
      const u16* Ax; int ldax;
      if (l == 0)      { Ax = Xe + (size_t)t * EMBD; ldax = Tmax * EMBD; }
      else if (l == 1) { Ax = Nb0; ldax = S0; }
      else             { Ax = Nb1; ldax = S1; }

      GemmArgs g{};
      g.Ax = Ax; g.ldax = ldax; g.K1 = ins[l];
      g.Ah = Sb + soff[l]; g.ldah = SSUM; g.K2 = outs[l];
      g.BT = KgT[l]; g.bias = bg[l];
      g.outs = outs[l]; g.ldh = SSUM;
      g.HfR = Hf + soff[l];
      g.Z = Zb; g.RHb = RHb;
      g.lens = lens; g.t = t;
      gru_gemm<0><<<(2 * outs[l]) / 128, 256, 0, stream>>>(g);

      GemmArgs c{};
      c.Ax = Ax; c.ldax = ldax; c.K1 = ins[l];
      c.Ah = RHb; c.ldah = outs[l]; c.K2 = outs[l];
      c.BT = KcT[l]; c.bias = bc[l];
      c.outs = outs[l]; c.ldh = SSUM;
      c.HfW = Hf + soff[l];
      c.Z = Zb;
      c.Sb = Sb + soff[l];
      c.Nb = (l == 0) ? Nb0 : (l == 1) ? Nb1 : nullptr;
      c.lens = lens; c.t = t;
      gru_gemm<1><<<outs[l] / 128, 256, 0, stream>>>(c);
    }
  }

  // Final projection: tanh(state @ W^T + b)
  GemmArgs f{};
  f.Ax = Sb; f.ldax = SSUM; f.K1 = SSUM;
  f.Ah = Sb; f.ldah = SSUM; f.K2 = 0;
  f.BT = WT; f.bias = bb;
  f.Out = Out;
  gru_gemm<2><<<LAT / 128, 256, 0, stream>>>(f);
}
